// NeuralODE_19181323944552
// MI455X (gfx1250) — matrix operand order repaired and checked by reading, compiled, not run on hardware
//
#include <hip/hip_runtime.h>

// ---------------------------------------------------------------------------
// Neural ODE (RK4, 99 steps) for MI455X / gfx1250, bf16 WMMA pipeline.
// Grid: 128 persistent blocks x 256 threads (8 waves). Each block owns 16
// batch rows and integrates them through all steps. Weights pre-packed into
// WMMA B-fragment layout (bf16) in d_ws by a prep kernel.
//
// NOTE: the asm-level memory clobber in vf_eval is essential: without it,
// LICM/GVN hoists all loop-invariant weight-fragment loads out of the step
// loop (they are identical across steps and RK4 stages), overflows the VGPR
// file and spills them to scratch -> every WMMA B operand becomes a scratch
// reload. The clobber keeps them as per-stage global_load_b128 from L2.
// ---------------------------------------------------------------------------

typedef __attribute__((ext_vector_type(16))) __bf16 v16bf;
typedef __attribute__((ext_vector_type(8)))  float  v8f;

#define WMMA_BF16(A, B, C) \
  __builtin_amdgcn_wmma_f32_16x16x32_bf16(false, (A), false, (B), (short)0, (C), false, false)

__device__ __forceinline__ unsigned short f32_to_bf16(float f) {
  union { float f; unsigned u; } v; v.f = f;
  unsigned r = v.u + 0x7FFFu + ((v.u >> 16) & 1u);   // round-to-nearest-even
  return (unsigned short)(r >> 16);
}

// Map k' (0..31, position within a 32-wide K tile) to A-fragment coords:
// VGPR index va (0..7), lane-half (0/1), element-within-pair (0/1).
__device__ __forceinline__ void kmap(int kp, int& va, int& half, int& el) {
  half = (kp >> 3) & 1;
  va   = ((kp & 16) ? 4 : 0) + ((kp & 7) >> 1);
  el   = kp & 1;
}

// ---------------------------------------------------------------------------
// Weight packing: f32 row-major -> bf16 B-fragment layout.
// B fragment (32x16 bf16): lane l, elem e  ->  k = 8*(l/16)+e%8+16*(e/8), n = l%16.
// Fragment region = 512 ushorts (32 lanes x 16 elems).
// w1p: 8 kt x 64 nt fragments (Wx, 256x1024). w2p: 32 kt x 16 nt (W2).
// ---------------------------------------------------------------------------
__global__ __launch_bounds__(256) void pack_weights_kernel(
    const float* __restrict__ W1, const float* __restrict__ W2,
    unsigned short* __restrict__ w1p, unsigned short* __restrict__ w2p) {
  int idx  = blockIdx.x * 256 + threadIdx.x;       // 0 .. 1024*512-1
  int le   = idx & 511;
  int f    = idx >> 9;
  int lane = le >> 4;
  int e    = le & 15;
  int kloc = ((lane >> 4) << 3) + (e & 7) + ((e & 8) << 1);   // 8*(lane/16)+e%8+16*(e/8)
  int n16  = lane & 15;
  if (f < 512) {                                   // W1 (Wx part): kt<8, nt<64
    int kt = f >> 6, nt = f & 63;
    int k  = 32 * kt + kloc;
    int n  = 16 * nt + n16;
    w1p[(size_t)f * 512 + le] = f32_to_bf16(W1[(size_t)k * 1024 + n]);
  } else {                                         // W2: kt<32, nt<16
    int f2 = f - 512;
    int kt = f2 >> 4, nt = f2 & 15;
    int k  = 32 * kt + kloc;
    int n  = 16 * nt + n16;
    w2p[(size_t)f2 * 512 + le] = f32_to_bf16(W2[(size_t)k * 256 + n]);
  }
}

// ---------------------------------------------------------------------------
// One vf evaluation: k = tanh(a@Wx + t*wt + b1) @ W2 + b2
// a, k are register C-fragments [2 col-tiles][8 vgpr] for this wave's 32 cols.
// ---------------------------------------------------------------------------
__device__ __forceinline__ void vf_eval(
    float t, const float (&a)[2][8], float (&k)[2][8],
    unsigned short* __restrict__ xA,      // [8][512]  A-frag layout, 16x256
    unsigned short* __restrict__ hA,      // [32][512] A-frag layout, 16x1024
    const unsigned short* __restrict__ w1p,
    const unsigned short* __restrict__ w2p,
    const float* __restrict__ b1v, const float* __restrict__ wt,
    const float* __restrict__ b2v,
    int wave, int lane, int lrow, int mhi) {

  // Barrier for the optimizer only: prevents weight-fragment loads from being
  // hoisted out of the 99-step loop / CSE'd across RK4 stages (spill bomb).
  asm volatile("" ::: "memory");

  // ---- scatter stage input (C-frag regs) into xA (A-frag layout) ----------
  // This wave owns cols 32*wave .. 32*wave+31  ->  K-tile kt == wave.
#pragma unroll
  for (int i = 0; i < 2; ++i) {
    int kp = 16 * i + lrow;                 // k' within the 32-wide K tile
    int va, half, el; kmap(kp, va, half, el);
    int eoff = 2 * va + el;
#pragma unroll
    for (int v = 0; v < 8; ++v) {
      int m     = v + 8 * mhi;              // row 0..15
      int laneA = m + 16 * half;
      xA[wave * 512 + laneA * 16 + eoff] = f32_to_bf16(a[i][v]);
    }
  }
  __syncthreads();

  // ---- GEMM1: h = tanh(a @ Wx + t*wt + b1), scatter h into hA -------------
  v16bf af[8];
#pragma unroll
  for (int kt = 0; kt < 8; ++kt)
    af[kt] = *(const v16bf*)(xA + kt * 512 + lane * 16);

  for (int j = 0; j < 8; ++j) {             // wave covers n-tiles 8w .. 8w+7
    int nt = wave * 8 + j;
    const unsigned short* wp = w1p + (size_t)nt * 512 + (size_t)lane * 16;
    if (j < 7)                               // prefetch next n-tile's fragments
      __builtin_prefetch(wp + 512, 0, 1);
    v8f acc = {0.f, 0.f, 0.f, 0.f, 0.f, 0.f, 0.f, 0.f};
#pragma unroll
    for (int kt = 0; kt < 8; ++kt) {
      v16bf bfrag = *(const v16bf*)(wp + (size_t)kt * 64 * 512);
      acc = WMMA_BF16(af[kt], bfrag, acc);
    }
    int   n  = nt * 16 + lrow;
    float bw = b1v[n] + t * wt[n];
    int va, half, el; kmap(n & 31, va, half, el);
    unsigned hbase = (unsigned)(n >> 5) * 512 + (unsigned)(2 * va + el);
#pragma unroll
    for (int v = 0; v < 8; ++v) {
      float hv  = tanhf(acc[v] + bw);
      int laneA = (v + 8 * mhi) + 16 * half;
      hA[hbase + laneA * 16] = f32_to_bf16(hv);
    }
  }
  __syncthreads();

  // ---- GEMM2: k = h @ W2 + b2 (wave covers out n-tiles 2w, 2w+1) ----------
  int nt0 = wave * 2, nt1 = nt0 + 1;
  v8f acc0 = {0.f, 0.f, 0.f, 0.f, 0.f, 0.f, 0.f, 0.f};
  v8f acc1 = {0.f, 0.f, 0.f, 0.f, 0.f, 0.f, 0.f, 0.f};
  for (int kt = 0; kt < 32; ++kt) {
    const unsigned short* wq0 = w2p + ((size_t)(kt * 16 + nt0)) * 512 + (size_t)lane * 16;
    if (kt < 31)                             // prefetch next K-tile's fragments
      __builtin_prefetch(wq0 + 16 * 512, 0, 1);
    v16bf ah  = *(const v16bf*)(hA + kt * 512 + lane * 16);
    v16bf bf0 = *(const v16bf*)(wq0);
    v16bf bf1 = *(const v16bf*)(wq0 + 512);
    acc0 = WMMA_BF16(ah, bf0, acc0);
    acc1 = WMMA_BF16(ah, bf1, acc1);
  }
  float bb0 = b2v[nt0 * 16 + lrow];
  float bb1 = b2v[nt1 * 16 + lrow];
#pragma unroll
  for (int v = 0; v < 8; ++v) { k[0][v] = acc0[v] + bb0; k[1][v] = acc1[v] + bb1; }
}

// ---------------------------------------------------------------------------
// Persistent RK4 kernel: block = 16 batch rows, 99 steps.
// ---------------------------------------------------------------------------
__global__ __launch_bounds__(256) void node_rk4_kernel(
    const float* __restrict__ x,
    const float* __restrict__ b1v,
    const float* __restrict__ wt,     // W1 row 256
    const float* __restrict__ b2v,
    const unsigned short* __restrict__ w1p,
    const unsigned short* __restrict__ w2p,
    float* __restrict__ out) {

  __shared__ __align__(16) unsigned short xA[8 * 512];    //  8 KB
  __shared__ __align__(16) unsigned short hA[32 * 512];   // 32 KB

  const int tid  = threadIdx.x;
  const int lane = tid & 31;
  const int wave = tid >> 5;
  const int lrow = lane & 15;
  const int mhi  = lane >> 4;
  const int bid  = blockIdx.x;

  const float dt = 1.0f / 99.0f;

  // Load xc (C-fragment layout) and emit traj[0] = x.
  float xc[2][8];
#pragma unroll
  for (int i = 0; i < 2; ++i)
#pragma unroll
    for (int v = 0; v < 8; ++v) {
      size_t off = ((size_t)bid * 16 + v + 8 * mhi) * 256
                 + (size_t)(32 * wave + 16 * i + lrow);
      xc[i][v] = x[off];
      out[off] = xc[i][v];
    }

  for (int s = 0; s < 99; ++s) {
    float t = (float)s * dt;
    float a[2][8], k[2][8], sum[2][8];

    // k1
    vf_eval(t, xc, k, xA, hA, w1p, w2p, b1v, wt, b2v, wave, lane, lrow, mhi);
#pragma unroll
    for (int i = 0; i < 2; ++i)
#pragma unroll
      for (int v = 0; v < 8; ++v) {
        sum[i][v] = k[i][v];
        a[i][v]   = xc[i][v] + 0.5f * dt * k[i][v];
      }
    // k2
    vf_eval(t + 0.5f * dt, a, k, xA, hA, w1p, w2p, b1v, wt, b2v, wave, lane, lrow, mhi);
#pragma unroll
    for (int i = 0; i < 2; ++i)
#pragma unroll
      for (int v = 0; v < 8; ++v) {
        sum[i][v] += 2.0f * k[i][v];
        a[i][v]    = xc[i][v] + 0.5f * dt * k[i][v];
      }
    // k3
    vf_eval(t + 0.5f * dt, a, k, xA, hA, w1p, w2p, b1v, wt, b2v, wave, lane, lrow, mhi);
#pragma unroll
    for (int i = 0; i < 2; ++i)
#pragma unroll
      for (int v = 0; v < 8; ++v) {
        sum[i][v] += 2.0f * k[i][v];
        a[i][v]    = xc[i][v] + dt * k[i][v];
      }
    // k4
    vf_eval(t + dt, a, k, xA, hA, w1p, w2p, b1v, wt, b2v, wave, lane, lrow, mhi);
#pragma unroll
    for (int i = 0; i < 2; ++i)
#pragma unroll
      for (int v = 0; v < 8; ++v) {
        sum[i][v] += k[i][v];
        xc[i][v]  += (dt / 6.0f) * sum[i][v];
      }

    // traj[s+1] = xc
    float* o = out + (size_t)(s + 1) * 2048 * 256;
#pragma unroll
    for (int i = 0; i < 2; ++i)
#pragma unroll
      for (int v = 0; v < 8; ++v) {
        size_t off = ((size_t)bid * 16 + v + 8 * mhi) * 256
                   + (size_t)(32 * wave + 16 * i + lrow);
        o[off] = xc[i][v];
      }
  }
}

// ---------------------------------------------------------------------------
extern "C" void kernel_launch(void* const* d_in, const int* in_sizes, int n_in,
                              void* d_out, int out_size, void* d_ws, size_t ws_size,
                              hipStream_t stream) {
  (void)in_sizes; (void)n_in; (void)out_size;
  const float* x  = (const float*)d_in[0];   // (2048, 256)
  const float* W1 = (const float*)d_in[1];   // (257, 1024); row 256 = wt
  const float* b1 = (const float*)d_in[2];   // (1024,)
  const float* W2 = (const float*)d_in[3];   // (1024, 256)
  const float* b2 = (const float*)d_in[4];   // (256,)
  float* out = (float*)d_out;                // (100, 2048, 256)

  unsigned short* w1p = (unsigned short*)d_ws;   // 512 frags * 512 u16 = 512 KB
  unsigned short* w2p = w1p + 512 * 512;         // 512 frags * 512 u16 = 512 KB
  if (ws_size < (size_t)2 * 512 * 512 * sizeof(unsigned short)) return;

  // Pack weights to bf16 WMMA B-fragment layout (1024 fragments total).
  pack_weights_kernel<<<dim3(2048), dim3(256), 0, stream>>>(W1, W2, w1p, w2p);

  // Persistent RK4 integration: 128 blocks x 16 rows.
  node_rk4_kernel<<<dim3(128), dim3(256), 0, stream>>>(
      x, b1, W1 + (size_t)256 * 1024, b2, w1p, w2p, out);
}
